// RecommendationModel_33277406609570
// MI455X (gfx1250) — matrix operand order repaired and checked by reading, not compiled, not run on hardware
//
#include <hip/hip_runtime.h>
#include <math.h>

typedef __attribute__((ext_vector_type(16))) _Float16 v16h;
typedef __attribute__((ext_vector_type(8)))  float    v8f;

#define C_DIM  128
#define HC_DIM 256

// ---------------------------------------------------------------- utilities

__global__ void k_fill_f32(float* __restrict__ p, float v, long n) {
  long i = (long)blockIdx.x * blockDim.x + threadIdx.x;
  if (i < n) p[i] = v;
}

// out[i,:] = table[ids[i],:]
__global__ void k_gather128(const float* __restrict__ table, const int* __restrict__ ids,
                            float* __restrict__ out, int nrows) {
  long gid = (long)blockIdx.x * blockDim.x + threadIdx.x;
  if (gid >= (long)nrows * C_DIM) return;
  long row = gid >> 7; int c = gid & 127;
  out[gid] = table[(long)ids[row] * C_DIM + c];
}

// nn.Embedding(max_norm=1.0): gather + L2 renorm per row (one wave32 per row)
__global__ void k_gather_renorm(const float* __restrict__ table, const int* __restrict__ ids,
                                float* __restrict__ out, int nrows) {
  int wave = (int)(((long)blockIdx.x * blockDim.x + threadIdx.x) >> 5);
  int lane = threadIdx.x & 31;
  if (wave >= nrows) return;
  const float* src = table + (long)ids[wave] * C_DIM;
  float v[4]; float ss = 0.f;
#pragma unroll
  for (int j = 0; j < 4; j++) { v[j] = src[lane + 32 * j]; ss += v[j] * v[j]; }
#pragma unroll
  for (int off = 16; off > 0; off >>= 1) ss += __shfl_xor(ss, off, 32);
  float nrm = sqrtf(ss);
  float sc = (nrm > 1.0f) ? (1.0f / (nrm + 1e-7f)) : 1.0f;
  float* dst = out + (long)wave * C_DIM;
#pragma unroll
  for (int j = 0; j < 4; j++) dst[lane + 32 * j] = v[j] * sc;
}

// r = recipe_x[NR,20] @ W[20,128] + b   (one thread per output element)
__global__ void k_linear20(const float* __restrict__ x, const float* __restrict__ W,
                           const float* __restrict__ b, float* __restrict__ y, int nrows) {
  long gid = (long)blockIdx.x * blockDim.x + threadIdx.x;
  if (gid >= (long)nrows * C_DIM) return;
  long row = gid >> 7; int c = gid & 127;
  const float* xr = x + row * 20;
  float s = b[c];
#pragma unroll
  for (int k = 0; k < 20; k++) s += xr[k] * W[k * C_DIM + c];
  y[gid] = s;
}

// column sums / sumsq for BatchNorm batch stats. stride keeps column fixed per thread.
__global__ void k_bn_reduce(const float* __restrict__ r, float* __restrict__ stats, int nrows) {
  int tid = threadIdx.x;            // 256
  int c = tid & 127;
  long total = (long)nrows * C_DIM;
  float s = 0.f, s2 = 0.f;
  for (long idx = (long)blockIdx.x * 256 + tid; idx < total; idx += (long)gridDim.x * 256) {
    float v = r[idx]; s += v; s2 += v * v;
  }
  __shared__ float ls[C_DIM], ls2[C_DIM];
  if (tid < C_DIM) { ls[tid] = 0.f; ls2[tid] = 0.f; }
  __syncthreads();
  atomicAdd(&ls[c], s); atomicAdd(&ls2[c], s2);
  __syncthreads();
  if (tid < C_DIM) { atomicAdd(&stats[tid], ls[tid]); atomicAdd(&stats[C_DIM + tid], ls2[tid]); }
}

__global__ void k_bn_apply(float* __restrict__ r, const float* __restrict__ stats,
                           const float* __restrict__ gamma, const float* __restrict__ beta,
                           int nrows) {
  long gid = (long)blockIdx.x * blockDim.x + threadIdx.x;
  if (gid >= (long)nrows * C_DIM) return;
  int c = gid & 127;
  float inv_n = 1.0f / (float)nrows;
  float mu = stats[c] * inv_n;
  float var = stats[C_DIM + c] * inv_n - mu * mu;
  r[gid] = gamma[c] * (r[gid] - mu) * rsqrtf(var + 1e-5f) + beta[c];
}

// wv[h*128+k] = sum_c Wdst[k, h*128+c] * attD[h,c]  (a_d collapses to a matvec)
__global__ void k_build_wv(const float* __restrict__ W, const float* __restrict__ att,
                           float* __restrict__ wv) {
  int tid = blockIdx.x * blockDim.x + threadIdx.x;
  if (tid >= 256) return;
  int h = tid >> 7, k = tid & 127;
  float s = 0.f;
  for (int c = 0; c < C_DIM; c++) s += W[(long)k * HC_DIM + h * C_DIM + c] * att[h * C_DIM + c];
  wv[tid] = s;
}

// a[i,h] = dot(x[i,:128], wv[h,:128])   (one wave per row, both heads)
__global__ void k_rowdot_ad(const float* __restrict__ x, const float* __restrict__ wv,
                            float* __restrict__ a, int nrows) {
  int wave = (int)(((long)blockIdx.x * blockDim.x + threadIdx.x) >> 5);
  int lane = threadIdx.x & 31;
  if (wave >= nrows) return;
  const float* xr = x + (long)wave * C_DIM;
  float s0 = 0.f, s1 = 0.f;
#pragma unroll
  for (int j = 0; j < 4; j++) {
    int idx = lane + 32 * j;
    float xv = xr[idx];
    s0 += xv * wv[idx];
    s1 += xv * wv[C_DIM + idx];
  }
#pragma unroll
  for (int off = 16; off > 0; off >>= 1) { s0 += __shfl_xor(s0, off, 32); s1 += __shfl_xor(s1, off, 32); }
  if (lane == 0) { a[(long)wave * 2 + 0] = s0; a[(long)wave * 2 + 1] = s1; }
}

// a_s[i,h] = sum_c hs[i, h*128+c] * att[h,c]   (one wave per row over 256 cols)
__global__ void k_rowdot_as(const float* __restrict__ hs, const float* __restrict__ att,
                            float* __restrict__ a, int nrows) {
  int wave = (int)(((long)blockIdx.x * blockDim.x + threadIdx.x) >> 5);
  int lane = threadIdx.x & 31;
  if (wave >= nrows) return;
  const float* hr = hs + (long)wave * HC_DIM;
  float s0 = 0.f, s1 = 0.f;
#pragma unroll
  for (int j = 0; j < 8; j++) {
    int idx = lane + 32 * j;
    float p = hr[idx] * att[idx];
    if (j < 4) s0 += p; else s1 += p;
  }
#pragma unroll
  for (int off = 16; off > 0; off >>= 1) { s0 += __shfl_xor(s0, off, 32); s1 += __shfl_xor(s1, off, 32); }
  if (lane == 0) { a[(long)wave * 2 + 0] = s0; a[(long)wave * 2 + 1] = s1; }
}

// ------------------------------------------------------- WMMA GEMM [N,128]@[128,256]
// Block = 256 thr (8 waves); block tile = 16 rows x 256 cols; each wave owns 32 cols
// (two 16x16 D tiles, A fragment reused). K=128 in 4 steps of 32 via
// v_wmma_f32_16x16x32_f16. LDS tiles are K-contiguous so every fragment
// half-pair is one aligned ds_load_b32.

union frag16 { v16h h; unsigned int u[8]; };

__device__ __forceinline__ unsigned int pack2h(float a, float b) {
  union { _Float16 h[2]; unsigned int u; } p;
  p.h[0] = (_Float16)a; p.h[1] = (_Float16)b;
  return p.u;                                  // lowers to v_cvt_pk_f16_f32
}

#define XPITCH 36   // halves per row (72B = 18 banks -> conflict-free fragment reads)
#define WPITCH 36

__global__ __launch_bounds__(256) void k_wmma_gemm(const float* __restrict__ X,
                                                   const float* __restrict__ W,
                                                   float* __restrict__ Y, int nrows) {
  __shared__ _Float16 Xs[16 * XPITCH];    // [row][k]    16 x 32 used
  __shared__ _Float16 Wt[256 * WPITCH];   // [col][k]   256 x 32 used (transposed tile)
  const int tid  = threadIdx.x;
  const int wave = tid >> 5;
  const int lane = tid & 31;
  const int rowBase = blockIdx.x * 16;

  // X staging assignment: thread -> (row rm, K-pair p)
  const int xrm = tid >> 4;            // 0..15
  const int xp  = tid & 15;            // pair index, k = 2*p
  int xrow = rowBase + xrm; if (xrow >= nrows) xrow = nrows - 1;
  const float* xptr = X + (long)xrow * C_DIM + 2 * xp;

  v8f acc0 = {}, acc1 = {};
#pragma unroll
  for (int kk = 0; kk < C_DIM; kk += 32) {
    if (kk) __syncthreads();           // protect LDS reuse across K steps
    // ---- stage X tile: one float2 load + one packed ds_store_b32 per thread
    {
      float2 xv = *(const float2*)(xptr + kk);
      *(unsigned int*)&Xs[xrm * XPITCH + 2 * xp] = pack2h(xv.x, xv.y);
    }
    // ---- stage W tile transposed: col = tid, 16 K-pairs per thread
#pragma unroll
    for (int p = 0; p < 16; p++) {
      int k = kk + 2 * p;
      float w0 = W[(long)k * HC_DIM + tid];
      float w1 = W[(long)(k + 1) * HC_DIM + tid];
      *(unsigned int*)&Wt[tid * WPITCH + 2 * p] = pack2h(w0, w1);
    }
    if (kk + 32 < C_DIM) {             // prefetch next K slab (global_prefetch_b8)
      __builtin_prefetch(&W[(long)(kk + 32) * HC_DIM + tid], 0, 1);
      __builtin_prefetch(xptr + kk + 32, 0, 1);
    }
    __syncthreads();

    // ---- A fragment: lane m=lane&15, hk=lane>>4, K = hk*16 + 2j
    frag16 a, b0, b1;
    const int m = lane & 15, hk = lane >> 4;
#pragma unroll
    for (int j = 0; j < 8; j++) {
      int k0 = hk * 16 + 2 * j;
      a.u[j] = *(const unsigned int*)&Xs[m * XPITCH + k0];
    }
    // ---- B fragments: n=lane&15, K = hk*16 + 2j (K-contiguous in Wt rows)
    const int c0 = wave * 32 + m;          // tile 0 column
#pragma unroll
    for (int j = 0; j < 8; j++) {
      int k0 = hk * 16 + 2 * j;
      b0.u[j] = *(const unsigned int*)&Wt[c0 * WPITCH + k0];
      b1.u[j] = *(const unsigned int*)&Wt[(c0 + 16) * WPITCH + k0];
    }
    acc0 = __builtin_amdgcn_wmma_f32_16x16x32_f16(false, a.h, false, b0.h, (short)0, acc0,
                                                  false, false);
    acc1 = __builtin_amdgcn_wmma_f32_16x16x32_f16(false, a.h, false, b1.h, (short)0, acc1,
                                                  false, false);
  }
  // ---- D layout: VGPR v -> M = (lane>>4)*8 + v, N = lane&15
  const int n = lane & 15, mh = lane >> 4;
#pragma unroll
  for (int v = 0; v < 8; v++) {
    int row = rowBase + mh * 8 + v;
    if (row < nrows) {
      Y[(long)row * HC_DIM + wave * 32 + n]      = acc0[v];
      Y[(long)row * HC_DIM + wave * 32 + 16 + n] = acc1[v];
    }
  }
}

// ---------------------------------------------------------------- edge phase

__device__ __forceinline__ void atomicMaxF(float* addr, float v) {
  if (v >= 0.f) atomicMax((int*)addr, __float_as_int(v));
  else          atomicMin((unsigned int*)addr, __float_as_uint(v));
}

__global__ void k_edge_logits_max(const int* __restrict__ src, const int* __restrict__ dst,
                                  int E, int nloop,
                                  const float* __restrict__ aS, const float* __restrict__ aD,
                                  float* __restrict__ e, float* __restrict__ m) {
  int idx = blockIdx.x * blockDim.x + threadIdx.x;
  int total = E + nloop;
  if (idx >= total) return;
  int s = (idx < E) ? src[idx] : (idx - E);
  int d = (idx < E) ? dst[idx] : (idx - E);
#pragma unroll
  for (int h = 0; h < 2; h++) {
    float v = aS[(long)s * 2 + h] + aD[(long)d * 2 + h];
    v = (v > 0.f) ? v : 0.2f * v;                      // leaky_relu(0.2)
    e[(long)idx * 2 + h] = v;
    atomicMaxF(&m[(long)d * 2 + h], v);
  }
}

__global__ void k_edge_exp_den(const int* __restrict__ dst, int E, int nloop,
                               const float* __restrict__ m,
                               float* __restrict__ e, float* __restrict__ den) {
  int idx = blockIdx.x * blockDim.x + threadIdx.x;
  int total = E + nloop;
  if (idx >= total) return;
  int d = (idx < E) ? dst[idx] : (idx - E);
#pragma unroll
  for (int h = 0; h < 2; h++) {
    float ex = __expf(e[(long)idx * 2 + h] - m[(long)d * 2 + h]);
    e[(long)idx * 2 + h] = ex;
    atomicAdd(&den[(long)d * 2 + h], ex);
  }
}

// one block (256 thr) per edge: acc[d, h*128+c] += hs[s, h*128+c] * alpha[h]
__global__ void k_edge_scatter(const int* __restrict__ src, const int* __restrict__ dst,
                               int E, int nloop,
                               const float* __restrict__ ex, const float* __restrict__ den,
                               const float* __restrict__ hs, float* __restrict__ acc) {
  int eidx = blockIdx.x;
  int tid = threadIdx.x;                 // 256
  int s = (eidx < E) ? src[eidx] : (eidx - E);
  int d = (eidx < E) ? dst[eidx] : (eidx - E);
  int h = tid >> 7;
  float alpha = ex[(long)eidx * 2 + h] / (den[(long)d * 2 + h] + 1e-16f);
  atomicAdd(&acc[(long)d * HC_DIM + tid], hs[(long)s * HC_DIM + tid] * alpha);
}

// out[i,c] = 0.5*(acc[i,c]+acc[i,128+c]) + bias[c] (+ resid[i,c])
__global__ void k_finalize(const float* __restrict__ acc, const float* __restrict__ bias,
                           const float* __restrict__ resid, float* __restrict__ out, int nrows) {
  long gid = (long)blockIdx.x * blockDim.x + threadIdx.x;
  if (gid >= (long)nrows * C_DIM) return;
  long row = gid >> 7; int c = gid & 127;
  float v = 0.5f * (acc[row * HC_DIM + c] + acc[row * HC_DIM + C_DIM + c]) + bias[c];
  if (resid) v += resid[gid];
  out[gid] = v;
}

// ---------------------------------------------------------------- host side

static void run_gat(const float* xsrc, int Ns, const float* xdst, int Nd,
                    const int* src, const int* dst, int E,
                    const float* Wsrc, const float* Wdst,
                    const float* attS, const float* attD, const float* bias,
                    const float* resid, float* outbuf,
                    float* hsBuf, float* aS, float* aD, float* mBuf, float* denBuf,
                    float* eBuf, float* wvBuf, float* accBuf, hipStream_t stream) {
  const int nloop = (Ns < Nd) ? Ns : Nd;
  const int total = E + nloop;

  k_build_wv<<<1, 256, 0, stream>>>(Wdst, attD, wvBuf);
  k_rowdot_ad<<<(Nd + 7) / 8, 256, 0, stream>>>(xdst, wvBuf, aD, Nd);

  k_wmma_gemm<<<(unsigned)((Ns + 15) / 16), 256, 0, stream>>>(xsrc, Wsrc, hsBuf, Ns);
  k_rowdot_as<<<(Ns + 7) / 8, 256, 0, stream>>>(hsBuf, attS, aS, Ns);

  long nd2 = (long)Nd * 2;
  k_fill_f32<<<(unsigned)((nd2 + 255) / 256), 256, 0, stream>>>(mBuf, -1e30f, nd2);
  k_fill_f32<<<(unsigned)((nd2 + 255) / 256), 256, 0, stream>>>(denBuf, 0.f, nd2);
  long ndc = (long)Nd * HC_DIM;
  k_fill_f32<<<(unsigned)((ndc + 255) / 256), 256, 0, stream>>>(accBuf, 0.f, ndc);

  k_edge_logits_max<<<(total + 255) / 256, 256, 0, stream>>>(src, dst, E, nloop, aS, aD, eBuf, mBuf);
  k_edge_exp_den<<<(total + 255) / 256, 256, 0, stream>>>(dst, E, nloop, mBuf, eBuf, denBuf);
  k_edge_scatter<<<total, 256, 0, stream>>>(src, dst, E, nloop, eBuf, denBuf, hsBuf, accBuf);

  long ndo = (long)Nd * C_DIM;
  k_finalize<<<(unsigned)((ndo + 255) / 256), 256, 0, stream>>>(accBuf, bias, resid, outbuf, Nd);
}

extern "C" void kernel_launch(void* const* d_in, const int* in_sizes, int n_in,
                              void* d_out, int out_size, void* d_ws, size_t ws_size,
                              hipStream_t stream) {
  const int* user_ids  = (const int*)d_in[0];
  const int* ing_ids   = (const int*)d_in[1];
  const float* recipe_x = (const float*)d_in[2];
  const int* ing_src = (const int*)d_in[3];
  const int* ing_dst = (const int*)d_in[4];
  const int* ub_src  = (const int*)d_in[5];
  const int* ub_dst  = (const int*)d_in[6];
  const float* user_table = (const float*)d_in[7];
  const float* ing_table  = (const float*)d_in[8];
  const float* W_rl = (const float*)d_in[9];
  const float* b_rl = (const float*)d_in[10];
  const float* bn_gamma = (const float*)d_in[11];
  const float* bn_beta  = (const float*)d_in[12];
  const float* g1_ws = (const float*)d_in[13];
  const float* g1_wd = (const float*)d_in[14];
  const float* g1_as = (const float*)d_in[15];
  const float* g1_ad = (const float*)d_in[16];
  const float* g1_b  = (const float*)d_in[17];
  const float* g2_ws = (const float*)d_in[18];
  const float* g2_wd = (const float*)d_in[19];
  const float* g2_as = (const float*)d_in[20];
  const float* g2_ad = (const float*)d_in[21];
  const float* g2_b  = (const float*)d_in[22];

  const int NU = in_sizes[0];
  const int NI = in_sizes[1];
  const int NR = in_sizes[2] / 20;
  const int E1 = in_sizes[3];
  const int E2 = in_sizes[5];
  const int maxNs = (NU > NI) ? NU : NI;
  const long maxE = ((long)E1 + NI > (long)E2 + NU) ? ((long)E1 + NI) : ((long)E2 + NU);

  float* ws = (float*)d_ws;
  size_t off = 0;
  float* uBuf   = ws + off; off += (size_t)NU * C_DIM;
  float* ingBuf = ws + off; off += (size_t)NI * C_DIM;
  float* rBuf   = ws + off; off += (size_t)NR * C_DIM;
  float* hsBuf  = ws + off; off += (size_t)maxNs * HC_DIM;
  float* accBuf = ws + off; off += (size_t)NR * HC_DIM;
  float* aS     = ws + off; off += (size_t)maxNs * 2;
  float* aD     = ws + off; off += (size_t)NR * 2;
  float* mBuf   = ws + off; off += (size_t)NR * 2;
  float* denBuf = ws + off; off += (size_t)NR * 2;
  float* eBuf   = ws + off; off += (size_t)maxE * 2;
  float* stats  = ws + off; off += 256;
  float* wvBuf  = ws + off; off += 256;

  // --- embeddings ---
  k_gather_renorm<<<(NU + 7) / 8, 256, 0, stream>>>(user_table, user_ids, uBuf, NU);
  {
    long n = (long)NI * C_DIM;
    k_gather128<<<(unsigned)((n + 255) / 256), 256, 0, stream>>>(ing_table, ing_ids, ingBuf, NI);
  }

  // --- recipe linear + BN (batch stats) ---
  {
    long n = (long)NR * C_DIM;
    k_linear20<<<(unsigned)((n + 255) / 256), 256, 0, stream>>>(recipe_x, W_rl, b_rl, rBuf, NR);
    k_fill_f32<<<1, 256, 0, stream>>>(stats, 0.f, 256);
    k_bn_reduce<<<512, 256, 0, stream>>>(rBuf, stats, NR);
    k_bn_apply<<<(unsigned)((n + 255) / 256), 256, 0, stream>>>(rBuf, stats, bn_gamma, bn_beta, NR);
  }

  // --- GAT1: ingredients -> recipes, residual into rBuf ---
  run_gat(ingBuf, NI, rBuf, NR, ing_src, ing_dst, E1,
          g1_ws, g1_wd, g1_as, g1_ad, g1_b,
          /*resid=*/rBuf, /*out=*/rBuf,
          hsBuf, aS, aD, mBuf, denBuf, eBuf, wvBuf, accBuf, stream);

  // --- GAT2: users -> recipes, writes final output ---
  run_gat(uBuf, NU, rBuf, NR, ub_src, ub_dst, E2,
          g2_ws, g2_wd, g2_as, g2_ad, g2_b,
          /*resid=*/nullptr, /*out=*/(float*)d_out,
          hsBuf, aS, aD, mBuf, denBuf, eBuf, wvBuf, accBuf, stream);
}
